// SimpleModel_15358803050979
// MI455X (gfx1250) — compile-verified, matrix operand order checked by reading, not run on hardware
//
#include <hip/hip_runtime.h>
#include <hip/hip_bf16.h>

// ---------------------------------------------------------------------------
// Problem sizes (fixed by the reference's setup_inputs)
// ---------------------------------------------------------------------------
#define BATCH 8192
#define DIN   4096
#define DOUT  4096

#define MT 128   // block tile M
#define NT 128   // block tile N
#define KT 32    // K step = one v_wmma_f32_16x16x32_f16
#define KITERS (DIN / KT)

typedef __attribute__((ext_vector_type(16))) _Float16 v16h;
typedef __attribute__((ext_vector_type(8)))  _Float16 v8h;
typedef __attribute__((ext_vector_type(4)))  _Float16 v4h;
typedef __attribute__((ext_vector_type(8)))  float    v8f;
typedef __attribute__((ext_vector_type(4)))  float    v4f;

// LDS row stride in halves: 32 data + 8 pad (80 B) -> conflict-free b128 reads
#define LDSK 40

__device__ __forceinline__ v8h cvt8(v4f a, v4f b) {
    v4h ha = __builtin_convertvector(a, v4h);
    v4h hb = __builtin_convertvector(b, v4h);
    return __builtin_shufflevector(ha, hb, 0, 1, 2, 3, 4, 5, 6, 7);
}

// Assemble a 16x32 f16 A/B fragment per the CDNA5 16-bit VGPR striping:
// this lane's K-octet at p[0..7] (VGPR0-3) and p[16..23] (VGPR4-7).
__device__ __forceinline__ v16h frag16(const _Float16* p) {
    v8h lo = *(const v8h*)p;
    v8h hi = *(const v8h*)(p + 16);
    return __builtin_shufflevector(lo, hi, 0, 1, 2, 3, 4, 5, 6, 7,
                                           8, 9, 10, 11, 12, 13, 14, 15);
}

// ---------------------------------------------------------------------------
// Kernel 0: zero the per-row |h| max accumulators (uint image of fp32 >= 0)
// ---------------------------------------------------------------------------
__global__ void rowmax_init_kernel(unsigned* __restrict__ rowmax) {
    int i = blockIdx.x * blockDim.x + threadIdx.x;
    if (i < BATCH) rowmax[i] = 0u;
}

// ---------------------------------------------------------------------------
// Kernel 1: h = x @ W^T + b  (f16 WMMA, f32 accumulate), fused row-|max|
// ---------------------------------------------------------------------------
__global__ __launch_bounds__(256)
void gemm_f16wmma_kernel(const float* __restrict__ X,
                         const float* __restrict__ W,
                         const float* __restrict__ bias,
                         float* __restrict__ H,
                         unsigned* __restrict__ rowmax) {
    __shared__ _Float16 sA[MT][LDSK];   // 10 KB, x tile   [m][k] f16
    __shared__ _Float16 sB[NT][LDSK];   // 10 KB, W tile   [n][k] f16 (B = W^T col = W row)

    const int tid   = threadIdx.x;
    const int lane  = tid & 31;
    const int wave  = tid >> 5;
    const int lmod  = lane & 15;     // column / row within 16-tile
    const int lhalf = lane >> 4;     // selects K-octet (A) / M-half (C)

    const int row0 = blockIdx.y * MT;    // batch rows
    const int col0 = blockIdx.x * NT;    // output cols

    // Loader mapping: thread t covers tile row t/2, K-halves [(t&1)*16, +16)
    const int lrow = tid >> 1;
    const int lk   = (tid & 1) * 16;

    const float* xptr = X + (size_t)(row0 + lrow) * DIN + lk;
    const float* wptr = W + (size_t)(col0 + lrow) * DIN + lk;

    // Register prefetch of the k=0 tiles (4x float4 each)
    v4f ax[4], aw[4];
#pragma unroll
    for (int i = 0; i < 4; ++i) {
        ax[i] = *(const v4f*)(xptr + i * 4);
        aw[i] = *(const v4f*)(wptr + i * 4);
    }

    // Wave grid: 2 (M) x 4 (N); each wave owns 64x32 = 4x2 WMMA tiles
    const int wm = wave >> 2;            // 0..1
    const int wn = wave & 3;             // 0..3

    v8f acc[4][2] = {};

    for (int kt = 0; kt < KITERS; ++kt) {
        __syncthreads();  // previous iteration's fragment reads complete

        // fp32 -> f16 convert and stage into LDS
        *(v8h*)&sA[lrow][lk]     = cvt8(ax[0], ax[1]);
        *(v8h*)&sA[lrow][lk + 8] = cvt8(ax[2], ax[3]);
        *(v8h*)&sB[lrow][lk]     = cvt8(aw[0], aw[1]);
        *(v8h*)&sB[lrow][lk + 8] = cvt8(aw[2], aw[3]);

        __syncthreads();  // tiles visible to all waves

        // Prefetch the next K tile while this one is consumed
        if (kt + 1 < KITERS) {
            xptr += KT;
            wptr += KT;
#pragma unroll
            for (int i = 0; i < 4; ++i) {
                ax[i] = *(const v4f*)(xptr + i * 4);
                aw[i] = *(const v4f*)(wptr + i * 4);
            }
        }

        // Fragments: A rows m = wm*64 + tm*16 + lmod; B cols n = wn*32 + tn*16 + lmod
        v16h afrag[4], bfrag[2];
#pragma unroll
        for (int tm = 0; tm < 4; ++tm)
            afrag[tm] = frag16(&sA[wm * 64 + tm * 16 + lmod][lhalf * 8]);
#pragma unroll
        for (int tn = 0; tn < 2; ++tn)
            bfrag[tn] = frag16(&sB[wn * 32 + tn * 16 + lmod][lhalf * 8]);

#pragma unroll
        for (int tm = 0; tm < 4; ++tm)
#pragma unroll
            for (int tn = 0; tn < 2; ++tn)
                acc[tm][tn] = __builtin_amdgcn_wmma_f32_16x16x32_f16(
                    /*neg_a=*/false, afrag[tm],
                    /*neg_b=*/false, bfrag[tn],
                    /*c_mod=*/(short)0, acc[tm][tn],
                    /*reuse_a=*/false, /*reuse_b=*/false);
    }

    // ---- Epilogue: + bias, store h (fp32), fused per-row |max| reduction ----
    const int gn0 = col0 + wn * 32 + lmod;
    const int gn1 = gn0 + 16;
    const float b0 = bias[gn0];
    const float b1 = bias[gn1];

#pragma unroll
    for (int tm = 0; tm < 4; ++tm) {
        const int mbase = row0 + wm * 64 + tm * 16 + lhalf * 8;
#pragma unroll
        for (int r = 0; r < 8; ++r) {
            const int gm = mbase + r;
            const float v0 = acc[tm][0][r] + b0;
            const float v1 = acc[tm][1][r] + b1;
            H[(size_t)gm * DOUT + gn0] = v0;
            H[(size_t)gm * DOUT + gn1] = v1;

            // max|h| over the 16 lanes holding this row (within wave half)
            float am = fmaxf(fabsf(v0), fabsf(v1));
            am = fmaxf(am, __shfl_xor(am, 1, 16));
            am = fmaxf(am, __shfl_xor(am, 2, 16));
            am = fmaxf(am, __shfl_xor(am, 4, 16));
            am = fmaxf(am, __shfl_xor(am, 8, 16));
            if (lmod == 0)
                atomicMax(&rowmax[gm], __float_as_uint(am));  // >=0: uint order == float order
        }
    }
}

// ---------------------------------------------------------------------------
// Kernel 2: in-place block quantize (shared row exponent, wl=8) + ReLU
//   step = 2^(floor(log2(max|row|)) - 6); q = floor(h/step + 0.5) * step
// ---------------------------------------------------------------------------
__global__ __launch_bounds__(256)
void quantize_relu_kernel(float* __restrict__ H,
                          const unsigned* __restrict__ rowmax) {
    const size_t i = ((size_t)blockIdx.x * blockDim.x + threadIdx.x) * 4;
    const int row = (int)(i >> 12);  // DOUT == 4096
    const float m = __uint_as_float(rowmax[row]);

    v4f v = *(const v4f*)(H + i);
    v4f o;
    if (m > 0.0f) {
        int e;
        (void)frexpf(m, &e);                    // m in [0.5,1)*2^e -> floor(log2 m) = e-1
        const float step = ldexpf(1.0f, e - 7); // 2^(e-1-6), exact
        const float inv  = ldexpf(1.0f, 7 - e); // exact reciprocal (power of two)
#pragma unroll
        for (int j = 0; j < 4; ++j)
            o[j] = fmaxf(floorf(v[j] * inv + 0.5f) * step, 0.0f);
    } else {
#pragma unroll
        for (int j = 0; j < 4; ++j)
            o[j] = fmaxf(v[j], 0.0f);           // all-zero row passthrough + relu
    }
    *(v4f*)(H + i) = o;
}

// ---------------------------------------------------------------------------
// Host-side launcher
// ---------------------------------------------------------------------------
extern "C" void kernel_launch(void* const* d_in, const int* in_sizes, int n_in,
                              void* d_out, int out_size, void* d_ws, size_t ws_size,
                              hipStream_t stream) {
    const float* x    = (const float*)d_in[0];  // [8192, 4096]
    const float* W    = (const float*)d_in[1];  // [4096, 4096]
    const float* bias = (const float*)d_in[2];  // [4096]
    float* out        = (float*)d_out;          // [8192, 4096] (holds h, then quantized)
    unsigned* rowmax  = (unsigned*)d_ws;        // [8192] uint images of row max|h|

    (void)in_sizes; (void)n_in; (void)out_size; (void)ws_size;

    // 1) zero the row-max accumulators (deterministic per call)
    rowmax_init_kernel<<<(BATCH + 255) / 256, 256, 0, stream>>>(rowmax);

    // 2) GEMM + bias -> h in d_out, fused row-|max| via atomics
    dim3 grid(DOUT / NT, BATCH / MT);  // (32, 64)
    gemm_f16wmma_kernel<<<grid, 256, 0, stream>>>(x, W, bias, out, rowmax);

    // 3) block quantize + relu, in place over d_out
    const size_t total = (size_t)BATCH * DOUT;          // 33,554,432
    quantize_relu_kernel<<<(unsigned)(total / 4 / 256), 256, 0, stream>>>(out, rowmax);
}
